// OuterProductMean_13675175870790
// MI455X (gfx1250) — matrix operand order repaired and checked by reading, not compiled, not run on hardware
//
#include <hip/hip_runtime.h>
#include <hip/hip_bf16.h>

typedef __attribute__((ext_vector_type(16))) __bf16 v16bf;
typedef __attribute__((ext_vector_type(8)))  float  v8f;

#define L_SEQ  768
#define C_IN   384
#define C_OUT  32
#define N_OUT  128

// workspace layout (bytes)
#define WS_LEFT_OFF   0                        // 768*32 f32 = 98304
#define WS_RIGHT_OFF  98304                    // 768*32 f32 = 98304
#define WS_WFRAG_OFF  196608                   // 8 dtiles * 2 kgrp * 32 lanes * 16 bf16 = 16384 B

// ---------------------------------------------------------------------------
// Kernel 1: LayerNorm + left/right projections (+mask, +bias). One block / row.
// ---------------------------------------------------------------------------
__global__ void ln_proj_kernel(const float* __restrict__ act,
                               const float* __restrict__ mask,
                               const float* __restrict__ gamma,
                               const float* __restrict__ beta,
                               const float* __restrict__ Wl,
                               const float* __restrict__ bl,
                               const float* __restrict__ Wr,
                               const float* __restrict__ br,
                               float* __restrict__ left,
                               float* __restrict__ right) {
    const int l = blockIdx.x;
    const int t = threadIdx.x;                  // 128 threads
    const float* row = act + (size_t)l * C_IN;

    __shared__ float red [128];
    __shared__ float red2[128];
    __shared__ float xn  [C_IN];

    float a0 = row[t], a1 = row[t + 128], a2 = row[t + 256];
    red [t] = a0 + a1 + a2;
    red2[t] = a0*a0 + a1*a1 + a2*a2;
    __syncthreads();
    for (int s = 64; s > 0; s >>= 1) {
        if (t < s) { red[t] += red[t + s]; red2[t] += red2[t + s]; }
        __syncthreads();
    }
    const float mu  = red[0]  * (1.0f / C_IN);
    const float var = red2[0] * (1.0f / C_IN) - mu * mu;
    const float rv  = rsqrtf(var + 1e-5f);

    xn[t]       = (a0 - mu) * rv * gamma[t]       + beta[t];
    xn[t + 128] = (a1 - mu) * rv * gamma[t + 128] + beta[t + 128];
    xn[t + 256] = (a2 - mu) * rv * gamma[t + 256] + beta[t + 256];
    __syncthreads();

    if (t < 64) {
        const int   d  = t & 31;
        const bool  rt = (t >= 32);
        const float* W = rt ? Wr : Wl;
        const float  b = rt ? br[d] : bl[d];
        float acc = 0.0f;
        #pragma unroll 8
        for (int c = 0; c < C_IN; ++c) acc += xn[c] * W[c * C_OUT + d];
        const float o = mask[l] * (acc + b);
        float* dst = rt ? right : left;
        dst[l * C_OUT + d] = o;
    }
}

// ---------------------------------------------------------------------------
// Kernel 2: swizzle W_out (64x128 f32) into per-lane bf16 B-fragments for
// v_wmma_f32_16x16x32_bf16:  lane n<16 -> col n, halves h = K 0..7,16..23;
//                            lane n>=16 -> col n-16, halves h = K 8..15,24..31.
// Fragment f = dtile*2 + kgrp (kgrp 0 = prod rows 0..31, 1 = diff rows 32..63).
// ---------------------------------------------------------------------------
__global__ void prep_wout_kernel(const float* __restrict__ W_out,
                                 __bf16* __restrict__ Wfrag) {
    const int e = blockIdx.x * 256 + threadIdx.x;   // 8192 elements
    if (e >= 8 * 2 * 32 * 16) return;
    const int h    = e & 15;
    const int lane = (e >> 4) & 31;
    const int g    = (e >> 9) & 1;
    const int tile = e >> 10;
    const int col  = tile * 16 + (lane & 15);
    const int krow = (h < 8) ? (((lane >> 4) << 3) + h) : (16 + ((lane >> 4) << 3) + h - 8);        // 0..31 within the half
    const int c    = g * 32 + krow;                 // row of W_out (2*C_OUT)
    Wfrag[e] = (__bf16)W_out[c * N_OUT + col];
}

// ---------------------------------------------------------------------------
// Kernel 3: fused pair-feature construction + GEMM with W_out.
// One wave per (i, j-tile of 16). 16 WMMAs/wave, 16x128 f32 output per wave.
// grid = (6, 768), block = 256 (8 waves); wave w -> jt = blockIdx.x*8 + w.
// ---------------------------------------------------------------------------
__global__ void pair_gemm_kernel(const float* __restrict__ left,
                                 const float* __restrict__ right,
                                 const __bf16* __restrict__ Wfrag,
                                 const float* __restrict__ b_out,
                                 float* __restrict__ out) {
    const int lane = threadIdx.x & 31;
    const int warp = threadIdx.x >> 5;
    const int i    = blockIdx.y;
    const int jt   = blockIdx.x * 8 + warp;
    const int j0   = jt * 16;

    const int grp  = lane >> 4;      // K half selector for A; M half for D
    const int colr = lane & 15;      // A row / D column
    const int j    = j0 + colr;
    const int a    = grp * 8;        // per-lane K base: {a..a+7} U {a+16..a+23}

    const float* Lp = left  + (size_t)j * C_OUT;
    const float* Rp = right + (size_t)i * C_OUT;

    // Build A-fragments for the two K-halves of the pair feature (K = 64):
    //   prod half: left[j,c] * right[i,c]   (c = 0..31)
    //   diff half: left[j,c] - right[i,c]   (c = 0..31 -> pair c 32..63)
    v16bf Ap, Ad;
    #pragma unroll
    for (int h = 0; h < 16; ++h) {
        const int c = a + ((h < 8) ? h : (h + 8));
        const float lv = Lp[c];
        const float rvv = Rp[c];
        Ap[h] = (__bf16)(lv * rvv);
        Ad[h] = (__bf16)(lv - rvv);
    }

    const v16bf* WF = (const v16bf*)Wfrag;
    float* obase = out + ((size_t)(i * L_SEQ + j0 + grp * 8)) * N_OUT + colr;

    #pragma unroll
    for (int t = 0; t < 8; ++t) {
        const v16bf Bp = WF[(t * 2 + 0) * 32 + lane];
        const v16bf Bd = WF[(t * 2 + 1) * 32 + lane];
        v8f acc = {};
        acc = __builtin_amdgcn_wmma_f32_16x16x32_bf16(
                  false, Ap, false, Bp, (short)0, acc, false, false);
        acc = __builtin_amdgcn_wmma_f32_16x16x32_bf16(
                  false, Ad, false, Bd, (short)0, acc, false, false);
        const float bb = b_out[t * 16 + colr];
        float* op = obase + t * 16;
        #pragma unroll
        for (int r = 0; r < 8; ++r)
            op[(size_t)r * N_OUT] = acc[r] + bb;   // D: VGPR r -> row (grp*8 + r)
    }
}

// ---------------------------------------------------------------------------
extern "C" void kernel_launch(void* const* d_in, const int* in_sizes, int n_in,
                              void* d_out, int out_size, void* d_ws, size_t ws_size,
                              hipStream_t stream) {
    (void)in_sizes; (void)n_in; (void)out_size; (void)ws_size;
    const float* act      = (const float*)d_in[0];
    const float* mask     = (const float*)d_in[1];
    const float* ln_gamma = (const float*)d_in[2];
    const float* ln_beta  = (const float*)d_in[3];
    const float* W_left   = (const float*)d_in[4];
    const float* b_left   = (const float*)d_in[5];
    const float* W_right  = (const float*)d_in[6];
    const float* b_right  = (const float*)d_in[7];
    const float* W_out    = (const float*)d_in[8];
    const float* b_out    = (const float*)d_in[9];
    float* out = (float*)d_out;

    float*  ws_left  = (float*)((char*)d_ws + WS_LEFT_OFF);
    float*  ws_right = (float*)((char*)d_ws + WS_RIGHT_OFF);
    __bf16* ws_wfrag = (__bf16*)((char*)d_ws + WS_WFRAG_OFF);

    ln_proj_kernel<<<dim3(L_SEQ), dim3(128), 0, stream>>>(
        act, mask, ln_gamma, ln_beta, W_left, b_left, W_right, b_right,
        ws_left, ws_right);

    prep_wout_kernel<<<dim3(32), dim3(256), 0, stream>>>(W_out, ws_wfrag);

    pair_gemm_kernel<<<dim3(6, L_SEQ), dim3(256), 0, stream>>>(
        ws_left, ws_right, ws_wfrag, b_out, out);
}
